// BooleanREWAHead_84293028151338
// MI455X (gfx1250) — 8-bit matrix operand order repaired and checked by reading, compiled, not run on hardware
//
#include <hip/hip_runtime.h>

typedef __attribute__((ext_vector_type(16))) __bf16 v16bf;
typedef __attribute__((ext_vector_type(8)))  float  v8f;
typedef __attribute__((ext_vector_type(8)))  int    v8i;

#define BN     2048
#define DD     1024
#define MBITS  256
#define DH     128
#define NBATCH 8

#define FUSED_LDS_BYTES (16 * BN * 2 + 32 * DH * 2)

template<int MODE>
__global__ void proj_kernel(const float* __restrict__ X, const float* __restrict__ W,
                            void* __restrict__ out, int C) {
  const int row0 = blockIdx.x * 16;
  const int colb = blockIdx.y * 64;
  const int t    = threadIdx.x;
  const int lane = t & 31;
  const int wave = t >> 5;
  const int half = lane >> 4;

  __shared__ __bf16 aS[16 * 32];
  __shared__ __bf16 bS[32 * 64];

  v8f acc = {};
  for (int k0 = 0; k0 < DD; k0 += 32) {
    {
      const int r  = t >> 3;
      const int c4 = (t & 7) * 4;
      const float4 v = *(const float4*)&X[(size_t)(row0 + r) * DD + k0 + c4];
      aS[r * 32 + c4 + 0] = (__bf16)v.x;
      aS[r * 32 + c4 + 1] = (__bf16)v.y;
      aS[r * 32 + c4 + 2] = (__bf16)v.z;
      aS[r * 32 + c4 + 3] = (__bf16)v.w;
    }
    #pragma unroll
    for (int i = 0; i < 4; ++i) {
      const int idx = t * 4 + i;
      const int r   = idx >> 4;
      const int c4  = (idx & 15) * 4;
      const float4 v = *(const float4*)&W[(size_t)(k0 + r) * C + colb + c4];
      bS[r * 64 + c4 + 0] = (__bf16)v.x;
      bS[r * 64 + c4 + 1] = (__bf16)v.y;
      bS[r * 64 + c4 + 2] = (__bf16)v.z;
      bS[r * 64 + c4 + 3] = (__bf16)v.w;
    }
    __syncthreads();

    v16bf a, bfrag;
    const int ar = lane & 15;
    #pragma unroll
    for (int j = 0; j < 8; ++j) {
      a[j]     = aS[ar * 32 + half * 8 + j];
      a[8 + j] = aS[ar * 32 + 16 + half * 8 + j];
    }
    const int bc = wave * 16 + (lane & 15);
    #pragma unroll
    for (int j = 0; j < 8; ++j) {
      bfrag[j]     = bS[(half * 8 + j) * 64 + bc];
      bfrag[8 + j] = bS[(16 + half * 8 + j) * 64 + bc];
    }
    acc = __builtin_amdgcn_wmma_f32_16x16x32_bf16(false, a, false, bfrag,
                                                  (short)0, acc, false, false);
    __syncthreads();
  }

  const int cn = colb + wave * 16 + (lane & 15);
  #pragma unroll
  for (int i = 0; i < 8; ++i) {
    const int r = row0 + i + half * 8;
    if (MODE == 0) {
      ((signed char*)out)[(size_t)r * C + cn] = acc[i] > 0.0f ? 1 : -1;
    } else {
      ((__bf16*)out)[(size_t)r * C + cn] = (__bf16)acc[i];
    }
  }
}

__global__ void fused_attn_kernel(const signed char* __restrict__ sq,
                                  const signed char* __restrict__ sk,
                                  const void* __restrict__ vpv,
                                  float* __restrict__ attn,
                                  float* __restrict__ out) {
  extern __shared__ char smem[];
  __bf16* sAttn = (__bf16*)smem;
  __bf16* bS    = (__bf16*)(smem + 16 * BN * 2);
  __shared__ float red[256];

  const __bf16* vp = (const __bf16*)vpv;
  const int row0 = blockIdx.x * 16;
  const int b    = blockIdx.y;
  const int t    = threadIdx.x;
  const int lane = t & 31;
  const int wave = t >> 5;
  const int half = lane >> 4;

  const signed char* aq = sq + ((size_t)b * BN + row0 + (lane & 15)) * MBITS;
  v8i afr[4];
  #pragma unroll
  for (int c = 0; c < 4; ++c) {
    const signed char* p = aq + c * 64;
    const int2 p0 = *(const int2*)(p + half * 8);
    const int2 p1 = *(const int2*)(p + 16 + half * 8);
    const int2 p2 = *(const int2*)(p + 32 + half * 8);
    const int2 p3 = *(const int2*)(p + 48 + half * 8);
    v8i tmp = {p0.x, p0.y, p1.x, p1.y, p2.x, p2.y, p3.x, p3.y};
    afr[c] = tmp;
  }
  for (int i = 0; i < 16; ++i) {
    const int n0 = (wave + 8 * i) * 16;
    const signed char* bk = sk + ((size_t)b * BN + n0 + (lane & 15)) * MBITS;
    v8i acc = {};
    #pragma unroll
    for (int c = 0; c < 4; ++c) {
      const signed char* p = bk + c * 64;
      const int2 q0 = *(const int2*)(p + half * 8);
      const int2 q1 = *(const int2*)(p + 16 + half * 8);
      const int2 q2 = *(const int2*)(p + 32 + half * 8);
      const int2 q3 = *(const int2*)(p + 48 + half * 8);
      v8i bb = {q0.x, q0.y, q1.x, q1.y, q2.x, q2.y, q3.x, q3.y};
      acc = __builtin_amdgcn_wmma_i32_16x16x64_iu8(true, afr[c], true, bb, acc,
                                                   false, false);
    }
    #pragma unroll
    for (int j = 0; j < 8; ++j) {
      const int r = j + half * 8;
      const float s = ((float)acc[j] + 256.0f) * 0.03125f;
      sAttn[r * BN + n0 + (lane & 15)] = (__bf16)s;
    }
  }
  __syncthreads();

  const int srow = t >> 4;
  const int sub  = t & 15;
  float mx = -3.0e38f;
  for (int c = 0; c < 32; ++c) {
    const int col = c * 64 + sub * 4;
    mx = fmaxf(mx, fmaxf(fmaxf((float)sAttn[srow * BN + col + 0],
                               (float)sAttn[srow * BN + col + 1]),
                         fmaxf((float)sAttn[srow * BN + col + 2],
                               (float)sAttn[srow * BN + col + 3])));
  }
  red[t] = mx;
  __syncthreads();
  if (sub < 8) red[t] = fmaxf(red[t], red[t + 8]);
  __syncthreads();
  if (sub < 4) red[t] = fmaxf(red[t], red[t + 4]);
  __syncthreads();
  if (sub < 2) red[t] = fmaxf(red[t], red[t + 2]);
  __syncthreads();
  if (sub < 1) red[t] = fmaxf(red[t], red[t + 1]);
  __syncthreads();
  mx = red[srow * 16];
  __syncthreads();

  float psum = 0.0f;
  for (int c = 0; c < 32; ++c) {
    const int col = c * 64 + sub * 4;
    psum += __expf((float)sAttn[srow * BN + col + 0] - mx);
    psum += __expf((float)sAttn[srow * BN + col + 1] - mx);
    psum += __expf((float)sAttn[srow * BN + col + 2] - mx);
    psum += __expf((float)sAttn[srow * BN + col + 3] - mx);
  }
  red[t] = psum;
  __syncthreads();
  if (sub < 8) red[t] += red[t + 8];
  __syncthreads();
  if (sub < 4) red[t] += red[t + 4];
  __syncthreads();
  if (sub < 2) red[t] += red[t + 2];
  __syncthreads();
  if (sub < 1) red[t] += red[t + 1];
  __syncthreads();
  const float inv = 1.0f / red[srow * 16];

  float* gRow = attn + ((size_t)b * BN + row0 + srow) * BN;
  for (int c = 0; c < 32; ++c) {
    const int col = c * 64 + sub * 4;
    float4 pv;
    pv.x = __expf((float)sAttn[srow * BN + col + 0] - mx) * inv;
    pv.y = __expf((float)sAttn[srow * BN + col + 1] - mx) * inv;
    pv.z = __expf((float)sAttn[srow * BN + col + 2] - mx) * inv;
    pv.w = __expf((float)sAttn[srow * BN + col + 3] - mx) * inv;
    *(float4*)&gRow[col] = pv;
    sAttn[srow * BN + col + 0] = (__bf16)pv.x;
    sAttn[srow * BN + col + 1] = (__bf16)pv.y;
    sAttn[srow * BN + col + 2] = (__bf16)pv.z;
    sAttn[srow * BN + col + 3] = (__bf16)pv.w;
  }
  __syncthreads();

  const __bf16* vb = vp + (size_t)b * BN * DH;
  v8f oacc = {};
  for (int k0 = 0; k0 < BN; k0 += 32) {
    {
      const int idx = t * 16;
      const int r = idx >> 7, c = idx & 127;
      const __bf16* g = &vb[(size_t)(k0 + r) * DH + c];
      const unsigned l = (unsigned)(size_t)(bS + r * DH + c);
      asm volatile("global_load_async_to_lds_b128 %0, %1, off\n\t"
                   "global_load_async_to_lds_b128 %2, %3, off"
                   :: "v"(l), "v"(g), "v"(l + 16u), "v"(g + 8)
                   : "memory");
      asm volatile("s_wait_asynccnt 0x0" ::: "memory");
    }
    __syncthreads();

    v16bf a, bb;
    const int ar = lane & 15;
    #pragma unroll
    for (int j = 0; j < 8; ++j) {
      a[j]     = sAttn[ar * BN + k0 + half * 8 + j];
      a[8 + j] = sAttn[ar * BN + k0 + 16 + half * 8 + j];
    }
    const int bc = wave * 16 + (lane & 15);
    #pragma unroll
    for (int j = 0; j < 8; ++j) {
      bb[j]     = bS[(half * 8 + j) * DH + bc];
      bb[8 + j] = bS[(16 + half * 8 + j) * DH + bc];
    }
    oacc = __builtin_amdgcn_wmma_f32_16x16x32_bf16(false, a, false, bb,
                                                   (short)0, oacc, false, false);
    __syncthreads();
  }

  #pragma unroll
  for (int j = 0; j < 8; ++j) {
    out[((size_t)b * BN + row0 + j + half * 8) * DH + wave * 16 + (lane & 15)] = oacc[j];
  }
}

extern "C" void kernel_launch(void* const* d_in, const int* in_sizes, int n_in,
                              void* d_out, int out_size, void* d_ws, size_t ws_size,
                              hipStream_t stream) {
  const float* Q  = (const float*)d_in[0];
  const float* K  = (const float*)d_in[1];
  const float* V  = (const float*)d_in[2];
  const float* Wq = (const float*)d_in[3];
  const float* Wk = (const float*)d_in[4];
  const float* Wv = (const float*)d_in[5];

  float* out  = (float*)d_out;
  float* attn = out + (size_t)NBATCH * BN * DH;

  signed char* sq = (signed char*)d_ws;
  signed char* sk = sq + (size_t)NBATCH * BN * MBITS;
  void*        vp = (void*)(sk + (size_t)NBATCH * BN * MBITS);

  const dim3 blk128(128);

  proj_kernel<0><<<dim3(NBATCH * BN / 16, MBITS / 64), blk128, 0, stream>>>(Q, Wq, sq, MBITS);
  proj_kernel<0><<<dim3(NBATCH * BN / 16, MBITS / 64), blk128, 0, stream>>>(K, Wk, sk, MBITS);
  proj_kernel<1><<<dim3(NBATCH * BN / 16, DH / 64),    blk128, 0, stream>>>(V, Wv, vp, DH);

  fused_attn_kernel<<<dim3(BN / 16, NBATCH), dim3(256), FUSED_LDS_BYTES, stream>>>(
      sq, sk, vp, attn, out);
}
